// HybridAttentionBlock_90580860272709
// MI455X (gfx1250) — matrix operand order repaired and checked by reading, not compiled, not run on hardware
//
#include <hip/hip_runtime.h>
#include <hip/hip_bf16.h>

// Problem constants (match reference)
#define BB      4
#define LQ_     512
#define HH      16
#define D_      128
#define MAXS    2048
#define BLKSZ   16
#define NBLKS   1024
#define TOTAL   (BB*LQ_)            // 2048
#define MBLK    (MAXS/BLKSZ)        // 128
#define SCALE_  0.08838834764831845f // 1/sqrt(128)

typedef __attribute__((ext_vector_type(16))) _Float16 v16h;
typedef __attribute__((ext_vector_type(8)))  _Float16 v8h;
typedef __attribute__((ext_vector_type(8)))  float    v8f;

union V16 { v16h v; v8h h[2]; };

// ---------------------------------------------------------------- stage 1: slot inverse map
__global__ void inv_init_kernel(int* __restrict__ inv, int n) {
    int i = blockIdx.x * blockDim.x + threadIdx.x;
    if (i < n) inv[i] = -1;
}
__global__ void inv_scatter_kernel(const int* __restrict__ slots, int* __restrict__ inv, int total) {
    int i = blockIdx.x * blockDim.x + threadIdx.x;
    if (i < total) inv[slots[i]] = i;
}

// ---------------------------------------------------------------- stage 2: RoPE(Q) -> f16 [B,H,LQ,D]
__global__ __launch_bounds__(256)
void prep_q_kernel(const float* __restrict__ Q,
                   const float* __restrict__ cosb,
                   const float* __restrict__ sinb,
                   _Float16* __restrict__ Qr) {
    int idx = blockIdx.x * blockDim.x + threadIdx.x;   // over TOTAL*H*D
    int d = idx % D_;
    int h = (idx / D_) % HH;
    int t = idx / (D_ * HH);                           // flat token
    size_t base = ((size_t)t * HH + h) * D_;
    float q   = Q[base + d];
    float rot = (d < D_/2) ? -Q[base + d + D_/2] : Q[base + d - D_/2];
    float c = cosb[(size_t)t * D_ + d];
    float s = sinb[(size_t)t * D_ + d];
    int b = t / LQ_, ql = t % LQ_;
    Qr[(((size_t)b * HH + h) * LQ_ + ql) * D_ + d] = (_Float16)(q * c + rot * s);
}

// ---------------------------------------------------------------- stage 3: gather + scatter-update K/V -> f16
// Kg: [B,H,MAXS,D]   Vt: [B,H,D,MAXS] (transposed for PV B-fragments)
__global__ __launch_bounds__(256)
void prep_kv_kernel(const float* __restrict__ K,  const float* __restrict__ V,
                    const float* __restrict__ Kc, const float* __restrict__ Vc,
                    const float* __restrict__ cosb, const float* __restrict__ sinb,
                    const int* __restrict__ btab, const int* __restrict__ inv,
                    _Float16* __restrict__ Kg, _Float16* __restrict__ Vt) {
    int idx = blockIdx.x * blockDim.x + threadIdx.x;   // over B*MAXS*H*D = 16.7M
    int d = idx % D_;
    int h = (idx / D_) % HH;
    int s = (idx / (D_ * HH)) % MAXS;
    int b = idx / (D_ * HH * MAXS);
    int token = btab[b * MBLK + (s >> 4)] * BLKSZ + (s & (BLKSZ - 1));
    int i = inv[token];
    float kv, vv;
    if (i >= 0) {
        size_t base = ((size_t)i * HH + h) * D_;
        float k   = K[base + d];
        float rot = (d < D_/2) ? -K[base + d + D_/2] : K[base + d - D_/2];
        kv = k * cosb[(size_t)i * D_ + d] + rot * sinb[(size_t)i * D_ + d];
        vv = V[base + d];
    } else {
        size_t base = ((size_t)token * HH + h) * D_;
        kv = Kc[base + d];
        vv = Vc[base + d];
    }
    size_t bh = (size_t)b * HH + h;
    Kg[(bh * MAXS + s) * D_ + d] = (_Float16)kv;
    Vt[(bh * D_ + d) * MAXS + s] = (_Float16)vv;
}

// ---------------------------------------------------------------- async copy helpers (CDNA5 ASYNCcnt path)
__device__ __forceinline__ void async_cp16(unsigned int lds_off, unsigned long long gaddr) {
    // GLOBAL_LOAD_ASYNC_TO_LDS_B128: per-lane 16B global -> LDS, tracked by ASYNCcnt
    asm volatile("global_load_async_to_lds_b128 %0, %1, off" :: "v"(lds_off), "v"(gaddr) : "memory");
}
__device__ __forceinline__ void wait_async_le8()  { asm volatile("s_wait_asynccnt 0x8" ::: "memory"); }
__device__ __forceinline__ void wait_async_zero() { asm volatile("s_wait_asynccnt 0x0" ::: "memory"); }

// ---------------------------------------------------------------- stage 4: flash attention with WMMA
// 128 threads = 4 waves; each wave owns one 16-query tile.
// K/V tiles for each 32-key chunk are staged once per block into double-buffered LDS
// via global_load_async_to_lds_b128 (overlapped with compute on the other buffer).
__global__ __launch_bounds__(128)
void attn_kernel(const _Float16* __restrict__ Qr,
                 const _Float16* __restrict__ Kg,
                 const _Float16* __restrict__ Vt,
                 const float* __restrict__ mask,
                 float* __restrict__ out) {
    __shared__ _Float16 Kt[2][32][D_];    // 2 x 8 KB : 32 key tokens x 128 dims
    __shared__ _Float16 Vl[2][D_][32];    // 2 x 8 KB : 128 dims x 32 key tokens
    __shared__ _Float16 Plds[4][16][32];  // per-wave P slab (C-layout -> A-layout re-striping)

    const int tid  = threadIdx.x;
    const int lane = tid & 31;
    const int wave = tid >> 5;
    const int hi   = lane >> 4;            // 0 | 1 : half-wave group
    const int lo   = lane & 15;
    const int qblk = blockIdx.x % (LQ_ / 64);
    const int bh   = blockIdx.x / (LQ_ / 64);
    const int b    = bh / HH;
    const int h    = bh % HH;
    const int q0   = (qblk * 4 + wave) * 16;   // local query base within (b,h)

    const _Float16* kbase = Kg + (size_t)bh * MAXS * D_;
    const _Float16* vbase = Vt + (size_t)bh * D_ * MAXS;

    // Per-thread async-copy segment assignments (8 x 16B per thread per chunk):
    // K tile: 512 segments of 16B; V tile: 512 segments of 16B; 128 threads x 4 each.
    int kseg_t[4], kseg_o[4], vseg_d[4], vseg_o[4];
    #pragma unroll
    for (int j = 0; j < 4; ++j) {
        int seg = tid + 128 * j;
        kseg_t[j] = seg >> 4;  kseg_o[j] = (seg & 15) * 8;   // row, half-offset
        vseg_d[j] = seg >> 2;  vseg_o[j] = (seg & 3) * 8;
    }

    // ---- Q A-fragments (A 16x32 layout: lane row = lo; contiguous K run = c*32 + hi*16 .. +15)
    V16 a[4];
    {
        const _Float16* qp = Qr + (((size_t)bh * LQ_) + (q0 + lo)) * D_;
        #pragma unroll
        for (int c = 0; c < 4; ++c) {
            a[c].h[0] = *(const v8h*)(qp + c*32 + hi*16);
            a[c].h[1] = *(const v8h*)(qp + c*32 + hi*16 + 8);
        }
    }

    v8f  o[8];
    float rmax[8], rsum[8];
    #pragma unroll
    for (int i = 0; i < 8; ++i) { o[i] = v8f{}; rmax[i] = -1e30f; rsum[i] = 0.0f; }

    // prologue: stage chunk 0 into buffer 0
    #pragma unroll
    for (int j = 0; j < 4; ++j) {
        async_cp16((unsigned int)(unsigned long long)&Kt[0][kseg_t[j]][kseg_o[j]],
                   (unsigned long long)(kbase + (size_t)kseg_t[j] * D_ + kseg_o[j]));
        async_cp16((unsigned int)(unsigned long long)&Vl[0][vseg_d[j]][vseg_o[j]],
                   (unsigned long long)(vbase + (size_t)vseg_d[j] * MAXS + vseg_o[j]));
    }

    for (int chunk = 0; chunk < MAXS; chunk += 32) {
        const int cur = (chunk >> 5) & 1;
        const int nxt = cur ^ 1;
        if (chunk + 32 < MAXS) {
            #pragma unroll
            for (int j = 0; j < 4; ++j) {
                async_cp16((unsigned int)(unsigned long long)&Kt[nxt][kseg_t[j]][kseg_o[j]],
                           (unsigned long long)(kbase + (size_t)(chunk + 32 + kseg_t[j]) * D_ + kseg_o[j]));
                async_cp16((unsigned int)(unsigned long long)&Vl[nxt][vseg_d[j]][vseg_o[j]],
                           (unsigned long long)(vbase + (size_t)vseg_d[j] * MAXS + chunk + 32 + vseg_o[j]));
            }
            wait_async_le8();   // in-order completion => current buffer's 8 are done
        } else {
            wait_async_zero();
        }
        __syncthreads();        // all waves' slices of buffer `cur` are resident

        // ---- scores: two 16x16 C tiles (keys chunk..+15 and chunk+16..+31)
        // B 32x16 layout: lane column = lo (key token), contiguous K run = c*32 + hi*16 .. +15
        v8f s0 = v8f{}, s1 = v8f{};
        #pragma unroll
        for (int c = 0; c < 4; ++c) {
            V16 b0, b1;
            b0.h[0] = *(const v8h*)(&Kt[cur][lo]     [c*32 + hi*16]);
            b0.h[1] = *(const v8h*)(&Kt[cur][lo]     [c*32 + hi*16 + 8]);
            b1.h[0] = *(const v8h*)(&Kt[cur][16 + lo][c*32 + hi*16]);
            b1.h[1] = *(const v8h*)(&Kt[cur][16 + lo][c*32 + hi*16 + 8]);
            s0 = __builtin_amdgcn_wmma_f32_16x16x32_f16(false, a[c].v, false, b0.v, (short)0, s0, false, false);
            s1 = __builtin_amdgcn_wmma_f32_16x16x32_f16(false, a[c].v, false, b1.v, (short)0, s1, false, false);
        }

        // ---- online softmax per row (row = r + 8*hi lives in element r, striped over 16 lanes)
        #pragma unroll
        for (int r = 0; r < 8; ++r) {
            const int row = r + 8 * hi;
            const float* mp = mask + (size_t)(q0 + row) * MAXS + chunk;
            float x0 = s0[r] * SCALE_ + mp[lo];
            float x1 = s1[r] * SCALE_ + mp[16 + lo];
            float cm = fmaxf(x0, x1);
            #pragma unroll
            for (int off = 8; off >= 1; off >>= 1) cm = fmaxf(cm, __shfl_xor(cm, off, 32));
            float nm = fmaxf(rmax[r], cm);
            float f  = __expf(rmax[r] - nm);
            rmax[r]  = nm;
            float p0 = __expf(x0 - nm);
            float p1 = __expf(x1 - nm);
            float ps = p0 + p1;
            #pragma unroll
            for (int off = 8; off >= 1; off >>= 1) ps += __shfl_xor(ps, off, 32);
            rsum[r] = rsum[r] * f + ps;
            #pragma unroll
            for (int nt = 0; nt < 8; ++nt) o[nt][r] *= f;
            Plds[wave][row][lo]      = (_Float16)p0;
            Plds[wave][row][16 + lo] = (_Float16)p1;
        }
        __syncthreads();
        // ---- P as A-fragment (lane row = lo; contiguous K run = hi*16 .. +15)
        V16 pa;
        pa.h[0] = *(const v8h*)(&Plds[wave][lo][hi*16]);
        pa.h[1] = *(const v8h*)(&Plds[wave][lo][hi*16 + 8]);

        // ---- O += P * V : B 32x16, lane column = d-offset (lo), contiguous K run over keys
        #pragma unroll
        for (int nt = 0; nt < 8; ++nt) {
            V16 vb;
            vb.h[0] = *(const v8h*)(&Vl[cur][nt*16 + lo][hi*16]);
            vb.h[1] = *(const v8h*)(&Vl[cur][nt*16 + lo][hi*16 + 8]);
            o[nt] = __builtin_amdgcn_wmma_f32_16x16x32_f16(false, pa.v, false, vb.v, (short)0, o[nt], false, false);
        }
        __syncthreads();  // all reads of buffer `cur` and P slab done before next stage/writes
    }

    // ---- normalize + store fp32 out[token, h, d]
    #pragma unroll
    for (int r = 0; r < 8; ++r) {
        const int row = r + 8 * hi;
        const float inv = 1.0f / rsum[r];
        const int tok = b * LQ_ + q0 + row;
        float* op = out + ((size_t)tok * HH + h) * D_;
        #pragma unroll
        for (int nt = 0; nt < 8; ++nt) op[nt*16 + lo] = o[nt][r] * inv;
    }
}

// ---------------------------------------------------------------- host launcher
extern "C" void kernel_launch(void* const* d_in, const int* in_sizes, int n_in,
                              void* d_out, int out_size, void* d_ws, size_t ws_size,
                              hipStream_t stream) {
    const float* Q      = (const float*)d_in[0];
    const float* K      = (const float*)d_in[1];
    const float* V      = (const float*)d_in[2];
    const float* Kcache = (const float*)d_in[3];
    const float* Vcache = (const float*)d_in[4];
    const float* cosb   = (const float*)d_in[5];
    const float* sinb   = (const float*)d_in[6];
    const float* mask   = (const float*)d_in[7];
    const int*   slots  = (const int*)d_in[10];
    const int*   btab   = (const int*)d_in[11];
    float* out = (float*)d_out;

    // workspace carve-up (all 256B-aligned sizes)
    const size_t QR_BYTES = (size_t)BB*HH*LQ_*D_ * sizeof(_Float16);   // 8 MB
    const size_t KG_BYTES = (size_t)BB*HH*MAXS*D_ * sizeof(_Float16);  // 32 MB
    const size_t VT_BYTES = KG_BYTES;                                   // 32 MB
    char* ws = (char*)d_ws;
    _Float16* Qr  = (_Float16*)(ws);
    _Float16* Kg  = (_Float16*)(ws + QR_BYTES);
    _Float16* Vt  = (_Float16*)(ws + QR_BYTES + KG_BYTES);
    int*      inv = (int*)     (ws + QR_BYTES + KG_BYTES + VT_BYTES);

    const int ncache = NBLKS * BLKSZ;                       // 16384
    inv_init_kernel<<<(ncache + 255)/256, 256, 0, stream>>>(inv, ncache);
    inv_scatter_kernel<<<(TOTAL + 255)/256, 256, 0, stream>>>(slots, inv, TOTAL);

    const int nq = TOTAL * HH * D_;                         // 4 Mi
    prep_q_kernel<<<nq/256, 256, 0, stream>>>(Q, cosb, sinb, Qr);

    const int nkv = BB * MAXS * HH * D_;                    // 16 Mi
    prep_kv_kernel<<<nkv/256, 256, 0, stream>>>(K, V, Kcache, Vcache, cosb, sinb, btab, inv, Kg, Vt);

    // B*H * (LQ/64) blocks, 4 waves each (one 16-query tile per wave)
    attn_kernel<<<BB*HH*(LQ_/64), 128, 0, stream>>>(Qr, Kg, Vt, mask, out);
}
